// DDT_30150670418584
// MI455X (gfx1250) — matrix operand order repaired and checked by reading, not compiled, not run on hardware
//
#include <hip/hip_runtime.h>

typedef __attribute__((ext_vector_type(16))) _Float16 v16h;
typedef __attribute__((ext_vector_type(8)))  float    v8f;
typedef __attribute__((ext_vector_type(4)))  _Float16 v4h;
typedef __attribute__((ext_vector_type(4)))  unsigned int v4u;
typedef __attribute__((ext_vector_type(8)))  int      v8i;
typedef __attribute__((ext_vector_type(4)))  int      v4i;

#define D_DIM        512
#define TILE_B       64
#define THREADS      128
#define XS_STRIDE    520      // halves, padded to avoid LDS bank conflicts
#define WT_STRIDE    520      // halves
#define DENSE_LEVELS 7        // levels 0..6 computed densely via WMMA
#define DENSE_NODES  127      // nodes 0..126 (slot 127 computed, never read)
#define ZS_STRIDE    128
#define NT_COUNT     8        // 8 N-tiles of 16 node columns
#define KC_COUNT     16       // 512 / 32 (K per f16 WMMA)
#define N_INTERNAL   1023
#define DEPTH_TOTAL  10

#if defined(__AMDGCN__) && __has_builtin(__builtin_amdgcn_tensor_load_to_lds) && \
    __has_builtin(__builtin_amdgcn_s_wait_tensorcnt)
#define USE_TDM 1
#else
#define USE_TDM 0
#endif

__global__ __launch_bounds__(THREADS, 1)
void ddt_fused_kernel(const float* __restrict__ X,
                      const float* __restrict__ W,
                      const float* __restrict__ Bb,
                      const float* __restrict__ LP,
                      float* __restrict__ Out,
                      float* __restrict__ OutLeaf)
{
    __shared__ _Float16 xs[TILE_B * XS_STRIDE];   // x tile, f16        (~66.5 KB)
    __shared__ _Float16 wt[16 * WT_STRIDE];       // W node-tile, f16   (~16.6 KB)
#if USE_TDM
    __shared__ float    wst[16 * D_DIM];          // TDM staging, f32   (32 KB)
#endif
    __shared__ float    zs[TILE_B * ZS_STRIDE];   // dense logits       (32 KB)

    const int tid  = threadIdx.x;
    const int lane = tid & 31;
    const int wave = tid >> 5;
    const int b0   = blockIdx.x * TILE_B;

    // ---------------- Phase A: stage x tile (f32 global -> f16 LDS) ----------
    for (int e = tid; e < TILE_B * (D_DIM / 4); e += THREADS) {
        const int r  = e >> 7;          // D_DIM/4 == 128
        const int k4 = e & 127;
        const float4 v = ((const float4*)X)[(size_t)(b0 + r) * (D_DIM / 4) + k4];
        v4h h;
        h[0] = (_Float16)v.x; h[1] = (_Float16)v.y;
        h[2] = (_Float16)v.z; h[3] = (_Float16)v.w;
        *(v4h*)(&xs[r * XS_STRIDE + k4 * 4]) = h;   // 8B-aligned (stride 1040B)
    }

#if USE_TDM
    // TDM descriptor builder: 1-D contiguous tile, 16 rows x 512 f32 = 8192 elems.
    // D# per ISA ch.8: group0 = {count|flags, lds_addr, gaddr_lo, gaddr_hi|type},
    // group1 = {data_size, tensor_dim0/1, tile_dim0/1/2, strides}.
    // This toolchain's builtin takes 6 args (g0, g1, g2, g3, g4, cpol).
    auto tdm_issue_wtile = [&](int nodeBase) {
        const unsigned long long ga =
            (unsigned long long)(const void*)(W + (size_t)nodeBase * D_DIM);
        const unsigned ldsOff = (unsigned)(unsigned long long)(void*)wst;
        v4u g0 = { 1u,                                   // count=1 (valid, user D#)
                   ldsOff,                               // lds_addr (bytes)
                   (unsigned)ga,                         // global_addr[31:0]
                   (unsigned)((ga >> 32) & 0x01FFFFFFu) | (2u << 30) }; // [56:32]|type=2
        v8i g1 = { (int)(2u << 16),        // data_size=2 (4B), wg_mask=0, no flags
                   (int)(8192u << 16),     // tensor_dim0[15:0]=8192 (atomic_bar_addr=0)
                   (int)(1u << 16),        // tensor_dim0[31:16]=0, tensor_dim1=1
                   (int)(8192u << 16),     // tensor_dim1[31:16]=0, tile_dim0=8192
                   1,                      // tile_dim1=1, tile_dim2=0
                   8192,                   // tensor_dim0_stride[31:0]
                   0, 0 };                 // stride0 hi / stride1
        v4i g2 = { 0, 0, 0, 0 };
        v4i g3 = { 0, 0, 0, 0 };
        v8i g4 = { 0, 0, 0, 0, 0, 0, 0, 0 };
        __builtin_amdgcn_tensor_load_to_lds(g0, g1, g2, g3, g4, 0);
    };
    if (tid < 32) tdm_issue_wtile(0);     // kick off tile 0 (wave 0 owns TENSORcnt)
#endif
    __syncthreads();

    // ---------------- Phase B: dense logits for nodes 0..127 via WMMA --------
    // A fragments: 16 samples per wave, full K=512 resident in VGPRs.
    // Layout (16-bit A, 16x32): lanes 0-15 hold K=[0..15] of row M=lane;
    // lanes 16-31 hold K=[16..31] of row M=lane-16 (matches B's contiguous halves).
    const int m0   = wave * 16;
    const int ksel = (lane >> 4) << 4;      // 0 or 16
    v16h afrag[KC_COUNT];
    {
        const _Float16* xrow = &xs[(m0 + (lane & 15)) * XS_STRIDE];
        #pragma unroll
        for (int kc = 0; kc < KC_COUNT; ++kc) {
            const int kb = kc * 32 + ksel;
            v16h t;
            #pragma unroll
            for (int i = 0; i < 8; ++i) {
                t[i]     = xrow[kb + i];
                t[i + 8] = xrow[kb + 8 + i];
            }
            afrag[kc] = t;
        }
    }

    // B layout (16-bit B, 32x16): lane n in 0-15 holds K=0..15 of column n,
    // lane n+16 holds K=16..31 of column n (contiguous halves per lane).
    const int ncol = lane & 15;
    const int koff = (lane >> 4) << 4;      // 0 or 16

    for (int nt = 0; nt < NT_COUNT; ++nt) {
        const int nodeBase = nt * 16;
#if USE_TDM
        if (tid < 32) __builtin_amdgcn_s_wait_tensorcnt(0);  // tile nt landed in wst
        __syncthreads();                 // publish wst; prior wt readers done
        for (int e = tid; e < 16 * D_DIM; e += THREADS) {    // convert f32 -> f16
            const int n = e >> 9;
            const int k = e & 511;
            wt[n * WT_STRIDE + k] = (_Float16)wst[e];
        }
        __syncthreads();                 // wt ready; wst free for next DMA
        if (tid < 32 && nt + 1 < NT_COUNT)
            tdm_issue_wtile(nodeBase + 16);   // overlap next DMA with WMMAs below
#else
        __syncthreads();
        for (int e = tid; e < 16 * D_DIM; e += THREADS) {
            const int n = e >> 9;
            const int k = e & 511;
            wt[n * WT_STRIDE + k] = (_Float16)W[(size_t)(nodeBase + n) * D_DIM + k];
        }
        __syncthreads();
#endif

        // Software-pipelined B-fragment feed: issue load(kc+1) before wmma(kc).
        const _Float16* wrow = &wt[ncol * WT_STRIDE + koff];
        v16h bcur;
        #pragma unroll
        for (int i = 0; i < 16; ++i) bcur[i] = wrow[i];

        v8f acc = {0.f, 0.f, 0.f, 0.f, 0.f, 0.f, 0.f, 0.f};
        #pragma unroll
        for (int kc = 0; kc < KC_COUNT; ++kc) {
            v16h bnext = bcur;
            if (kc + 1 < KC_COUNT) {
                #pragma unroll
                for (int i = 0; i < 16; ++i) bnext[i] = wrow[(kc + 1) * 32 + i];
            }
            acc = __builtin_amdgcn_wmma_f32_16x16x32_f16(
                      false, afrag[kc], false, bcur,
                      (short)0, acc, false, false);
            bcur = bnext;
        }
        // C layout: VGPR r -> M = (lane>>4)*8 + r, N = lane&15
        #pragma unroll
        for (int r = 0; r < 8; ++r) {
            const int m = ((lane >> 4) << 3) + r;
            zs[(m0 + m) * ZS_STRIDE + nodeBase + ncol] = acc[r];
        }
    }
    __syncthreads();

    // ---------------- Phase C: routing (2 threads per sample) ----------------
    const int s    = tid >> 1;
    const int hsel = tid & 1;
    const int gb   = b0 + s;

    int   idx  = 0;
    float gacc = 1.0f;

    #pragma unroll
    for (int lvl = 0; lvl < DENSE_LEVELS; ++lvl) {
        const float z   = zs[s * ZS_STRIDE + idx] + Bb[idx];
        const float val = 1.0f / (1.0f + __expf(-z));
        const int right = (z >= 0.0f) ? 1 : 0;   // sigmoid(z) >= 0.5  <=>  z >= 0
        gacc *= right ? val : (1.0f - val);
        idx = 2 * idx + 1 + right;
    }

    // levels 7..9: per-sample gather dot; W row is L2-resident (2 MB total)
    const _Float16* xrowc = &xs[s * XS_STRIDE + hsel * 256];
    __builtin_prefetch(&W[(size_t)idx * D_DIM + hsel * 256], 0, 1);
    for (int lvl = DENSE_LEVELS; lvl < DEPTH_TOTAL; ++lvl) {
        const float* wrowg = &W[(size_t)idx * D_DIM + hsel * 256];
        float p = 0.0f;
        #pragma unroll 8
        for (int k = 0; k < 256; ++k) p += (float)xrowc[k] * wrowg[k];
        p += __shfl_xor(p, 1, 32);               // pair reduction (lanes 2s, 2s+1)
        const float z   = p + Bb[idx];
        const float val = 1.0f / (1.0f + __expf(-z));
        const int right = (z >= 0.0f) ? 1 : 0;
        gacc *= right ? val : (1.0f - val);
        idx = 2 * idx + 1 + right;
        if (lvl + 1 < DEPTH_TOTAL)
            __builtin_prefetch(&W[(size_t)idx * D_DIM + hsel * 256], 0, 1);
    }

    const int leaf = idx - N_INTERNAL;
    const float* lp = &LP[(size_t)leaf * 64 + hsel * 32];
    float*       op = &Out[(size_t)gb * 64 + hsel * 32];
    #pragma unroll
    for (int j = 0; j < 32; ++j) op[j] = gacc * lp[j];
    if (hsel == 0) OutLeaf[gb] = (float)leaf;
}

extern "C" void kernel_launch(void* const* d_in, const int* in_sizes, int n_in,
                              void* d_out, int out_size, void* d_ws, size_t ws_size,
                              hipStream_t stream) {
    const float* X  = (const float*)d_in[0];   // [B, 512]
    const float* W  = (const float*)d_in[1];   // [1023, 512]
    const float* Bb = (const float*)d_in[2];   // [1023]
    const float* LP = (const float*)d_in[3];   // [1024, 64]
    // d_in[4] = max_depth (== 10, compiled in)

    const int Btot = in_sizes[0] / D_DIM;
    float* Out     = (float*)d_out;                       // [B, 64]
    float* OutLeaf = Out + (size_t)Btot * 64;             // [B] leaf idx (as float)

    const int grid = Btot / TILE_B;                       // 2048 workgroups
    ddt_fused_kernel<<<grid, THREADS, 0, stream>>>(X, W, Bb, LP, Out, OutLeaf);
}
